// ISTA_58016418234810
// MI455X (gfx1250) — matrix operand order repaired and checked by reading, not compiled, not run on hardware
//
#include <hip/hip_runtime.h>

// ---------------------------------------------------------------------------
// ISTA deconvolution for MI455X (gfx1250, wave32, WMMA).
//
// update = corr(y - corr(y,h), h_rev) is loop-invariant; the T=5 ISTA loop is
// a pointwise recurrence on it. Two bf16-WMMA Toeplitz-GEMM conv passes with
// fused epilogues; f32 accumulation via v_wmma_f32_16x16x32_bf16.
// Toeplitz B fragments are precomputed once (in wave32 fragment layout) by a
// tiny setup kernel and broadcast-loaded from L2 by every wave.
// Wide 256-column strips per block cut halo read amplification to 1.56x.
// ---------------------------------------------------------------------------

typedef __attribute__((ext_vector_type(16))) __bf16 v16bf;
typedef __attribute__((ext_vector_type(8)))  float  v8f;

#define BDIM   4096
#define LDIM   4096
#define KTAPS  128
#define PADL   63                       // (K-1)/2, TF-SAME left pad
#define TILE   16
#define WAVES  8                        // waves per block (256 threads)
#define TPW    2                        // column tiles per wave
#define COLS_PER_BLOCK (TILE * WAVES * TPW)   // 256 output columns / block
#define KSTEPS 5                        // ceil((16+128-1)/32) = 5 K-steps of 32
#define STAGE_W  (COLS_PER_BLOCK + 144) // 400 staged columns (covers t=0..159 of last tile)
#define STAGE_WP 404                    // padded row stride (u16): mult of 4 -> 8B-aligned
                                        // quad gathers; 202-dword stride bank-conflict free
                                        // for 16-row column access (10r % 64 distinct, r<16)
#define LAMBDA 0.1f
#define TITER  5
#define BMAT_ELEMS (2 * KSTEPS * 32 * 16)   // both passes' B fragments, u16 each

static __device__ __forceinline__ unsigned short f32_to_bf16_bits(float f) {
    unsigned int u = __builtin_bit_cast(unsigned int, f);
    u += 0x7fffu + ((u >> 16) & 1u);            // round-to-nearest-even
    return (unsigned short)(u >> 16);
}
static __device__ __forceinline__ __bf16 bits_to_bf16(unsigned short v) {
    return __builtin_bit_cast(__bf16, v);
}

// ---------------------------------------------------------------------------
// Setup: materialize Toeplitz B fragments for both passes in the contiguous-K wave32
// B-matrix layout.  Bt[K][N] = h[K-N] (pass0) or h_rev[K-N] (pass1).
// B order used here (contiguous K): lanes 0-15 hold K = 32s + e   (N = lane),
//                            lanes 16-31 hold K = 32s+16+e (N = lane-16).
// Storage index: ((pass*KSTEPS + s)*32 + lane)*16 + e   -> per-lane 32B blocks.
// ---------------------------------------------------------------------------
__global__ __launch_bounds__(256)
void build_toeplitz(const float* __restrict__ h, unsigned short* __restrict__ bmat)
{
    for (int idx = threadIdx.x; idx < BMAT_ELEMS; idx += 256) {
        const int e    = idx & 15;
        const int lane = (idx >> 4) & 31;
        const int s    = (idx >> 9) % KSTEPS;
        const int pass = idx / (KSTEPS * 32 * 16);
        const int K = 32 * s + 16 * (lane >> 4) + e;
        const int d = K - (lane & 15);
        float hv = 0.0f;
        if (d >= 0 && d < KTAPS) hv = pass ? h[KTAPS - 1 - d] : h[d];
        bmat[idx] = f32_to_bf16_bits(hv);
    }
}

// PASS2=false: rout(bf16) = y - corr(y, h)            (reads yf32)
// PASS2=true : xout(f32)  = ISTA_5(corr(r, h_rev))    (reads rbf16)
template <bool PASS2>
__global__ __launch_bounds__(WAVES * 32)
void ista_conv_wmma(const float* __restrict__ yf32,
                    const unsigned short* __restrict__ rbf16,
                    const unsigned short* __restrict__ bmat,
                    const float* __restrict__ step,
                    unsigned short* __restrict__ rout,
                    float* __restrict__ xout)
{
    __shared__ alignas(16) unsigned short smem[TILE * STAGE_WP];

    const int tid    = threadIdx.x;
    const int c0     = blockIdx.x * COLS_PER_BLOCK;   // first output column of block
    const int b0     = blockIdx.y * TILE;             // first batch row of block
    const int cstart = c0 - PADL;                     // leftmost staged input column

    // ---- Stage 16 x 400 bf16 strip of the input into LDS (zero-padded) ----
    for (int idx = tid; idx < TILE * STAGE_W; idx += WAVES * 32) {
        const int row  = idx / STAGE_W;
        const int col  = idx - row * STAGE_W;
        const int gcol = cstart + col;
        unsigned short v;
        if constexpr (PASS2) {
            v = (gcol >= 0 && gcol < LDIM)
                    ? rbf16[(size_t)(b0 + row) * LDIM + gcol] : (unsigned short)0;
        } else {
            const float f = (gcol >= 0 && gcol < LDIM)
                    ? yf32[(size_t)(b0 + row) * LDIM + gcol] : 0.0f;
            v = f32_to_bf16_bits(f);
        }
        smem[row * STAGE_WP + col] = v;
    }
    __syncthreads();

    const int lane  = tid & 31;
    const int wave  = tid >> 5;
    const int hi    = lane >> 4;     // half-wave select (K-subset for A/B, M-subset for C/D)
    const int nrow  = lane & 15;     // M for A-frags, N for B/C/D-frags

    // ---- Load precomputed Toeplitz B fragments (L2 broadcast, 32B/lane) ---
    const unsigned short* bbase =
        bmat + (size_t)(PASS2 ? KSTEPS : 0) * 32 * 16 + (size_t)lane * 16;
    v16bf bfrag[KSTEPS];
#pragma unroll
    for (int s = 0; s < KSTEPS; ++s)
        bfrag[s] = *(const v16bf*)(bbase + (size_t)s * 32 * 16);

    const float s0 = PASS2 ? step[0] : 0.0f;

    // ---- Each wave computes TPW 16x16 output tiles from the shared strip --
#pragma unroll
    for (int j = 0; j < TPW; ++j) {
        const int tilecol = wave * TPW + j;
        const int i0loc   = tilecol * TILE;   // tile origin inside staged strip

        // K-loop: 5 x v_wmma_f32_16x16x32_bf16, f32 accumulate
        v8f acc = {};
#pragma unroll
        for (int s = 0; s < KSTEPS; ++s) {
            v16bf afrag;
            // A layout (matches B fragments): element e holds K = 16*hi + e; M=lane%16.
#pragma unroll
            for (int e = 0; e < 16; ++e) {
                const int k = 16 * hi + e;
                const int t = 32 * s + k;               // window position 0..159
                afrag[e] = bits_to_bf16(smem[nrow * STAGE_WP + i0loc + t]);
            }
            acc = __builtin_amdgcn_wmma_f32_16x16x32_bf16(
                      false, afrag, false, bfrag[s], (short)0, acc, false, false);
        }

        // Fused epilogues (C/D layout: M = r + 8*hi, N = lane%16)
        const size_t colg = (size_t)(c0 + tilecol * TILE + nrow);
        if constexpr (!PASS2) {
#pragma unroll
            for (int r = 0; r < 8; ++r) {
                const size_t off = (size_t)(b0 + r + 8 * hi) * LDIM + colg;
                rout[off] = f32_to_bf16_bits(yf32[off] - acc[r]);
            }
        } else {
#pragma unroll
            for (int r = 0; r < 8; ++r) {
                const size_t off = (size_t)(b0 + r + 8 * hi) * LDIM + colg;
                const float u = acc[r];
                float x = 0.0f;
#pragma unroll
                for (int it = 0; it < TITER; ++it) {
                    const float t2 = x + s0 * u;
                    x = copysignf(fmaxf(fabsf(t2) - LAMBDA, 0.0f), t2);
                }
                xout[off] = x;
            }
        }
    }
}

extern "C" void kernel_launch(void* const* d_in, const int* in_sizes, int n_in,
                              void* d_out, int out_size, void* d_ws, size_t ws_size,
                              hipStream_t stream) {
    const float* y    = (const float*)d_in[0];   // [4096, 4096] f32
    const float* h    = (const float*)d_in[1];   // [128] f32
    const float* step = (const float*)d_in[2];   // [1] f32

    // Workspace layout: [0, 32MB) bf16 residual; then 10KB of B fragments.
    unsigned short* rbuf = (unsigned short*)d_ws;
    unsigned short* bmat = rbuf + (size_t)BDIM * LDIM;
    float* xout = (float*)d_out;

    dim3 grid(LDIM / COLS_PER_BLOCK, BDIM / TILE);  // (16, 256)
    dim3 block(WAVES * 32);                         // 256 threads = 8 wave32

    // Pass 0: build both passes' Toeplitz B fragments (10 KB, stays in L2).
    build_toeplitz<<<dim3(1), dim3(256), 0, stream>>>(h, bmat);
    // Pass 1: r = bf16(y - corr(y, h))
    ista_conv_wmma<false><<<grid, block, 0, stream>>>(y, nullptr, bmat, step, rbuf, nullptr);
    // Pass 2: x = ISTA_5(corr(r, h_rev))
    ista_conv_wmma<true ><<<grid, block, 0, stream>>>(nullptr, rbuf, bmat, step, nullptr, xout);
}
